// GNN_34256659153095
// MI455X (gfx1250) — matrix operand order repaired and checked by reading, not compiled, not run on hardware
//
#include <hip/hip_runtime.h>
#include <hip/hip_bf16.h>

typedef __attribute__((ext_vector_type(16))) __bf16 v16bf;
typedef __attribute__((ext_vector_type(4)))  __bf16 v4bf;
typedef __attribute__((ext_vector_type(8)))  float  v8f;

#define NNODE 256
#define CCH   512

// Identity within each 32-channel group: an A-fragment (per-lane
// k = 16h..16h+15) is 16 contiguous bf16, matching the row-major B operand.
static __device__ __forceinline__ int swz32(int w) {
    return w;
}

// ---------------------------------------------------------------------------
// Plain f32 -> bf16 conversion (row-major kept; used for B operands/weights).
// ---------------------------------------------------------------------------
__global__ __launch_bounds__(256)
void cvt_bf16_kernel(const float* __restrict__ src, __bf16* __restrict__ dst, int n) {
    for (int t = blockIdx.x * 256 + threadIdx.x; t * 4 < n; t += gridDim.x * 256) {
        const float4 v = *(const float4*)&src[t * 4];
        v4bf o;
        o[0] = (__bf16)v.x; o[1] = (__bf16)v.y; o[2] = (__bf16)v.z; o[3] = (__bf16)v.w;
        *(v4bf*)&dst[t * 4] = o;
    }
}

// ---------------------------------------------------------------------------
// f32 -> bf16 with A-fragment group swizzle (used for x / x_mid A operands).
// ---------------------------------------------------------------------------
__global__ __launch_bounds__(256)
void cvt_bf16_swz_kernel(const float* __restrict__ src, __bf16* __restrict__ dst, int n) {
    for (int t = blockIdx.x * 256 + threadIdx.x; t * 4 < n; t += gridDim.x * 256) {
        const int c = t * 4;                 // 4-aligned, never straddles an 8-block
        const float4 v = *(const float4*)&src[c];
        v4bf o;
        o[0] = (__bf16)v.x; o[1] = (__bf16)v.y; o[2] = (__bf16)v.z; o[3] = (__bf16)v.w;
        const int d = (c & ~31) + swz32(c & 31);
        *(v4bf*)&dst[d] = o;
    }
}

// ---------------------------------------------------------------------------
// P = X @ W^T using bf16 WMMA.  Xsw: [256][512] bf16 (group-swizzled),
// Wbf: [512][512] bf16 row-major, P: [256][512] f32.
// grid = (32 n-tiles, 16 m-tiles), block = 32 (one wave per 16x16 C tile)
// ---------------------------------------------------------------------------
__global__ __launch_bounds__(32)
void proj_gemm_kernel(const __bf16* __restrict__ Xsw,
                      const __bf16* __restrict__ Wbf,
                      float* __restrict__ P) {
    const int lane = threadIdx.x & 31;
    const int n0 = blockIdx.x * 16;
    const int m0 = blockIdx.y * 16;

    const int am = lane & 15, ah = lane >> 4;
    const int bn = lane & 15, bh = lane >> 4;

    const __bf16* arow = Xsw + (m0 + am) * CCH + ah * 16;
    const __bf16* brow = Wbf + (n0 + bn) * CCH + bh * 16;

    v8f acc = {};
#pragma unroll 4
    for (int kt = 0; kt < 16; ++kt) {
        const v16bf af = *(const v16bf*)(arow + kt * 32);
        const v16bf bf = *(const v16bf*)(brow + kt * 32);
        acc = __builtin_amdgcn_wmma_f32_16x16x32_bf16(false, af, false, bf,
                                                      (short)0, acc, false, false);
    }
    const int cn = n0 + (lane & 15);
#pragma unroll
    for (int r = 0; r < 8; ++r) {
        const int cm = m0 + r + 8 * (lane >> 4);
        P[cm * CCH + cn] = acc[r];
    }
}

// ---------------------------------------------------------------------------
// Fused edge pipeline for one 32-edge tile (fixed i, 32 consecutive j):
//   e = edgeRows @ E^T ; pre = e + Vix[i] + Vjx[j] ; row-BN ; relu ;
//   edge_out = edge_in + relu ; w = exp(sigmoid(edge_out)) ;
//   atomic S1[i,c] += w, S2[i,c] += w*Ujx[j,c]
// Two M-tiles per wave: each B fragment is loaded once and feeds two WMMAs,
// halving the L2 B-operand traffic (which otherwise rivals the HBM floor).
// grid = 2048 (65536/32), block = 32 (one wave).
// ---------------------------------------------------------------------------
#define AT_STRIDE 528   // bf16/row: rows stay 32B-aligned, banks rotate by 8
#define ET_STRIDE 516   // f32/row:  rows stay 16B-aligned, banks rotate by 4
#define MROWS     32    // edges per block

__global__ __launch_bounds__(32)
void edge_layer_kernel(const float* __restrict__ edge_in,
                       float* __restrict__ edge_out,
                       const __bf16* __restrict__ Ebf,   // [512][512] bf16 row-major
                       const float* __restrict__ Vix,
                       const float* __restrict__ Vjx,
                       const float* __restrict__ Ujx,
                       float* __restrict__ S1,
                       float* __restrict__ S2) {
    __shared__ __bf16 aT[MROWS * AT_STRIDE];
    __shared__ float  eT[MROWS * ET_STRIDE];
    __shared__ float  rowMean[MROWS], rowRstd[MROWS];

    const int lane = threadIdx.x & 31;
    const int tile = blockIdx.x;
    const int i  = tile >> 3;                 // 8 tiles of 32 edges per node i
    const int j0 = (tile & 7) * MROWS;
    const int ebase = (i * NNODE + j0) * CCH;

    // Stage 32 edge rows into LDS as bf16, pre-swizzled into A-fragment order.
    for (int t = lane; t < (MROWS * CCH) / 4; t += 32) {
        const int idx = t * 4;
        const int r = idx >> 9, c = idx & (CCH - 1);
        const float4 v = *(const float4*)&edge_in[ebase + r * CCH + c];
        v4bf o;
        o[0] = (__bf16)v.x; o[1] = (__bf16)v.y; o[2] = (__bf16)v.z; o[3] = (__bf16)v.w;
        *(v4bf*)&aT[r * AT_STRIDE + (c & ~31) + swz32(c & 31)] = o;
    }

    const int am = lane & 15, ah = lane >> 4;
    const int bn = lane & 15, bh = lane >> 4;
    const __bf16* arow0 = &aT[am * AT_STRIDE + ah * 16];                  // rows 0..15
    const __bf16* arow1 = &aT[(16 + am) * AT_STRIDE + ah * 16];           // rows 16..31

    for (int nt = 0; nt < 32; ++nt) {
        const int n0 = nt * 16;
        const __bf16* brow = Ebf + (n0 + bn) * CCH + bh * 16;
        v8f acc0 = {}, acc1 = {};
#pragma unroll 4
        for (int kt = 0; kt < 16; ++kt) {
            const v16bf bf  = *(const v16bf*)(brow + kt * 32);            // loaded once
            const v16bf af0 = *(const v16bf*)(arow0 + kt * 32);
            const v16bf af1 = *(const v16bf*)(arow1 + kt * 32);
            acc0 = __builtin_amdgcn_wmma_f32_16x16x32_bf16(false, af0, false, bf,
                                                           (short)0, acc0, false, false);
            acc1 = __builtin_amdgcn_wmma_f32_16x16x32_bf16(false, af1, false, bf,
                                                           (short)0, acc1, false, false);
        }
        const int n = n0 + (lane & 15);
        const float vix = Vix[i * CCH + n];
#pragma unroll
        for (int r = 0; r < 8; ++r) {
            const int m = r + 8 * (lane >> 4);
            eT[m * ET_STRIDE + n]        = acc0[r] + vix + Vjx[(j0 + m) * CCH + n];
            eT[(16 + m) * ET_STRIDE + n] = acc1[r] + vix + Vjx[(j0 + 16 + m) * CCH + n];
        }
    }

    // Per-row mean / rstd over 512 channels (half row per lane, pair via shfl).
#pragma unroll
    for (int rr = 0; rr < 2; ++rr) {
        const int r = rr * 16 + (lane & 15);
        const int half = lane >> 4;
        const float* row = &eT[r * ET_STRIDE + half * 256];
        float s = 0.f, ss = 0.f;
#pragma unroll 4
        for (int c = 0; c < 256; c += 4) {
            const float4 v = *(const float4*)&row[c];
            s  += v.x + v.y + v.z + v.w;
            ss += v.x * v.x + v.y * v.y + v.z * v.z + v.w * v.w;
        }
        s  += __shfl_xor(s, 16, 32);
        ss += __shfl_xor(ss, 16, 32);
        const float mean = s * (1.f / 512.f);
        const float var  = ss * (1.f / 512.f) - mean * mean;
        if (half == 0) { rowMean[r] = mean; rowRstd[r] = rsqrtf(var + 1e-5f); }
    }

    // BN -> relu -> residual edge write, and S1/S2 partials in registers.
    float s1r[16], s2r[16];
#pragma unroll
    for (int t = 0; t < 16; ++t) { s1r[t] = 0.f; s2r[t] = 0.f; }

    for (int m = 0; m < MROWS; ++m) {
        const int j = j0 + m;
        const float mean = rowMean[m], rstd = rowRstd[m];
#pragma unroll
        for (int t = 0; t < 16; ++t) {
            const int c = lane + 32 * t;              // coalesced 128B per step
            const float pre = eT[m * ET_STRIDE + c];
            float b = (pre - mean) * rstd;
            b = b > 0.f ? b : 0.f;
            const float ne = edge_in[ebase + m * CCH + c] + b;
            edge_out[ebase + m * CCH + c] = ne;
            const float sig = 1.f / (1.f + __expf(-ne));
            const float w = __expf(sig);
            s1r[t] += w;
            s2r[t] += w * Ujx[j * CCH + c];
        }
    }
#pragma unroll
    for (int t = 0; t < 16; ++t) {
        const int c = lane + 32 * t;
        atomicAdd(&S1[i * CCH + c], s1r[t]);
        atomicAdd(&S2[i * CCH + c], s2r[t]);
    }
}

// ---------------------------------------------------------------------------
// x_out = relu(x_res + bn_row(xU + (S2/S1)/256))   (per-node BN over 512 ch)
// grid = 256 (node), block = 128
// ---------------------------------------------------------------------------
__global__ __launch_bounds__(128)
void x_update_kernel(const float* __restrict__ x_res,
                     const float* __restrict__ xU,
                     const float* __restrict__ S1,
                     const float* __restrict__ S2,
                     float* __restrict__ x_out) {
    __shared__ float row[CCH];
    __shared__ float reds[4], redss[4];
    const int m = blockIdx.x;
    const int tid = threadIdx.x;

    float s = 0.f, ss = 0.f;
    for (int c = tid; c < CCH; c += 128) {
        const float agg = (S2[m * CCH + c] / S1[m * CCH + c]) * (1.f / 256.f);
        const float xn = xU[m * CCH + c] + agg;
        row[c] = xn;
        s += xn; ss += xn * xn;
    }
#pragma unroll
    for (int off = 16; off > 0; off >>= 1) {
        s  += __shfl_xor(s,  off, 32);
        ss += __shfl_xor(ss, off, 32);
    }
    if ((tid & 31) == 0) { reds[tid >> 5] = s; redss[tid >> 5] = ss; }
    __syncthreads();
    const float S  = reds[0] + reds[1] + reds[2] + reds[3];
    const float SS = redss[0] + redss[1] + redss[2] + redss[3];
    const float mean = S * (1.f / 512.f);
    const float var  = SS * (1.f / 512.f) - mean * mean;
    const float rstd = rsqrtf(var + 1e-5f);
    for (int c = tid; c < CCH; c += 128) {
        const float b = (row[c] - mean) * rstd;
        const float v = x_res[m * CCH + c] + b;
        x_out[m * CCH + c] = v > 0.f ? v : 0.f;
    }
}

// ---------------------------------------------------------------------------
extern "C" void kernel_launch(void* const* d_in, const int* in_sizes, int n_in,
                              void* d_out, int out_size, void* d_ws, size_t ws_size,
                              hipStream_t stream) {
    (void)in_sizes; (void)n_in; (void)out_size; (void)ws_size;

    const float* x    = (const float*)d_in[0];
    const float* edge = (const float*)d_in[1];
    const float* U1 = (const float*)d_in[2];
    const float* V1 = (const float*)d_in[3];
    const float* A1 = (const float*)d_in[4];
    const float* B1 = (const float*)d_in[5];
    const float* E1 = (const float*)d_in[6];
    const float* U2 = (const float*)d_in[7];
    const float* V2 = (const float*)d_in[8];
    const float* A2 = (const float*)d_in[9];
    const float* B2 = (const float*)d_in[10];
    const float* E2 = (const float*)d_in[11];

    float* out_x    = (float*)d_out;                 // [256*512]
    float* out_edge = out_x + NNODE * CCH;           // [65536*512]

    const int NC = NNODE * CCH;                      // 131072
    const int WW = CCH * CCH;                        // 262144

    float* ws    = (float*)d_ws;
    float* Vix   = ws;
    float* Vjx   = Vix + NC;
    float* Ujx   = Vjx + NC;
    float* xU    = Ujx + NC;
    float* S1    = xU  + NC;
    float* S2    = S1  + NC;                         // contiguous with S1
    float* x_mid = S2  + NC;
    __bf16* Xbf  = (__bf16*)(x_mid + NC);            // [256*512]  swizzled A operand
    __bf16* Abf  = Xbf + NC;                         // 5 weight B operands
    __bf16* Bbf  = Abf + WW;
    __bf16* Vbf  = Bbf + WW;
    __bf16* Ubf  = Vbf + WW;
    __bf16* Ebf  = Ubf + WW;

    const dim3 pgrid(32, 16), pblk(32);
    const int cvtBlkW = WW / (4 * 256);              // 256 blocks for a 512x512 matrix
    const int cvtBlkX = NC / (4 * 256);              // 128 blocks for x

    // ---- layer 1 ----
    cvt_bf16_swz_kernel<<<cvtBlkX, 256, 0, stream>>>(x, Xbf, NC);
    cvt_bf16_kernel<<<cvtBlkW, 256, 0, stream>>>(A1, Abf, WW);
    cvt_bf16_kernel<<<cvtBlkW, 256, 0, stream>>>(B1, Bbf, WW);
    cvt_bf16_kernel<<<cvtBlkW, 256, 0, stream>>>(V1, Vbf, WW);
    cvt_bf16_kernel<<<cvtBlkW, 256, 0, stream>>>(U1, Ubf, WW);
    cvt_bf16_kernel<<<cvtBlkW, 256, 0, stream>>>(E1, Ebf, WW);
    proj_gemm_kernel<<<pgrid, pblk, 0, stream>>>(Xbf, Abf, Vix);
    proj_gemm_kernel<<<pgrid, pblk, 0, stream>>>(Xbf, Bbf, Vjx);
    proj_gemm_kernel<<<pgrid, pblk, 0, stream>>>(Xbf, Vbf, Ujx);
    proj_gemm_kernel<<<pgrid, pblk, 0, stream>>>(Xbf, Ubf, xU);
    (void)hipMemsetAsync(S1, 0, (size_t)2 * NC * sizeof(float), stream);
    edge_layer_kernel<<<2048, 32, 0, stream>>>(edge, out_edge, Ebf, Vix, Vjx, Ujx, S1, S2);
    x_update_kernel<<<NNODE, 128, 0, stream>>>(x, xU, S1, S2, x_mid);

    // ---- layer 2 (edge updated in place in d_out) ----
    cvt_bf16_swz_kernel<<<cvtBlkX, 256, 0, stream>>>(x_mid, Xbf, NC);
    cvt_bf16_kernel<<<cvtBlkW, 256, 0, stream>>>(A2, Abf, WW);
    cvt_bf16_kernel<<<cvtBlkW, 256, 0, stream>>>(B2, Bbf, WW);
    cvt_bf16_kernel<<<cvtBlkW, 256, 0, stream>>>(V2, Vbf, WW);
    cvt_bf16_kernel<<<cvtBlkW, 256, 0, stream>>>(U2, Ubf, WW);
    cvt_bf16_kernel<<<cvtBlkW, 256, 0, stream>>>(E2, Ebf, WW);
    proj_gemm_kernel<<<pgrid, pblk, 0, stream>>>(Xbf, Abf, Vix);
    proj_gemm_kernel<<<pgrid, pblk, 0, stream>>>(Xbf, Bbf, Vjx);
    proj_gemm_kernel<<<pgrid, pblk, 0, stream>>>(Xbf, Vbf, Ujx);
    proj_gemm_kernel<<<pgrid, pblk, 0, stream>>>(Xbf, Ubf, xU);
    (void)hipMemsetAsync(S1, 0, (size_t)2 * NC * sizeof(float), stream);
    edge_layer_kernel<<<2048, 32, 0, stream>>>(out_edge, out_edge, Ebf, Vix, Vjx, Ujx, S1, S2);
    x_update_kernel<<<NNODE, 128, 0, stream>>>(x_mid, xU, S1, S2, out_x);
}
